// BaseFConv2d_15350213116004
// MI455X (gfx1250) — matrix operand order repaired and checked by reading, not compiled, not run on hardware
//
#include <hip/hip_runtime.h>

typedef __attribute__((ext_vector_type(16))) _Float16 v16h;
typedef __attribute__((ext_vector_type(8)))  _Float16 v8h;
typedef __attribute__((ext_vector_type(8)))  float    v8f;

#define B_     256
#define C_     3
#define L_     8
#define OC_    64
#define NNODES 7
#define INW    3072          // C*H*W
#define NPIX2  256           // 16*16 pooled pixels
#define K1PAD  32            // 27 padded to one WMMA K chunk
#define FCIN   (OC_*NPIX2)   // 16384
#define OUTW   10

// ---------------- weight conversion ----------------
// conv1: [l*64+oc][27] -> f16 [row][32] (K padded)
__global__ __launch_bounds__(256) void k_convert_w1(const float* __restrict__ w1,
                                                    _Float16* __restrict__ wf) {
    int idx = blockIdx.x * 256 + threadIdx.x;
    if (idx >= 512 * K1PAD) return;
    int row = idx / K1PAD, kk = idx % K1PAD;
    float v = (kk < 27) ? w1[row * 27 + kk] : 0.0f;
    wf[idx] = (_Float16)v;
}

// conv2: [l][oc][ic][r] -> f16 [l][r][oc][ic]  (ic innermost: contiguous A K-chunks)
__global__ __launch_bounds__(256) void k_convert_w2(const float* __restrict__ w2,
                                                    _Float16* __restrict__ wf) {
    int idx = blockIdx.x * 256 + threadIdx.x;          // 8*9*64*64 = 294912
    if (idx >= L_ * 9 * OC_ * OC_) return;
    int ic = idx & 63;
    int oc = (idx >> 6) & 63;
    int r  = (idx >> 12) % 9;
    int l  = idx / (9 * 64 * 64);
    wf[idx] = (_Float16)w2[((l * OC_ + oc) * OC_ + ic) * 9 + r];
}

// ---------------- soft-tree mixture: [B, 8] ----------------
__global__ __launch_bounds__(256) void k_mixture(const float* __restrict__ x,
                                                 const float* __restrict__ nw,
                                                 const float* __restrict__ nb,
                                                 float* __restrict__ mix) {
    __shared__ float sP[NNODES * 256];
    __shared__ float sLog[NNODES];
    int b = blockIdx.x, tid = threadIdx.x;
    float p[NNODES];
#pragma unroll
    for (int n = 0; n < NNODES; ++n) p[n] = 0.0f;
    for (int i = tid; i < INW; i += 256) {
        float xv = x[b * INW + i];
#pragma unroll
        for (int n = 0; n < NNODES; ++n) p[n] += xv * nw[n * INW + i];
    }
#pragma unroll
    for (int n = 0; n < NNODES; ++n) sP[n * 256 + tid] = p[n];
    __syncthreads();
    if (tid < NNODES) {
        float s = 0.0f;
        for (int t = 0; t < 256; ++t) s += sP[tid * 256 + t];
        sLog[tid] = s + nb[tid];
    }
    __syncthreads();
    if (tid == 0) {
        float sg[NNODES];
#pragma unroll
        for (int n = 0; n < NNODES; ++n) sg[n] = 1.0f / (1.0f + __expf(-sLog[n]));
#pragma unroll
        for (int j = 0; j < L_; ++j) {
            float t0 = ((j >> 2) & 1) ? sg[0] : 1.0f - sg[0];
            int   n1 = 1 + ((j >> 2) & 1);
            float t1 = ((j >> 1) & 1) ? sg[n1] : 1.0f - sg[n1];
            int   n2 = 3 + (j >> 1);
            float t2 = (j & 1) ? sg[n2] : 1.0f - sg[n2];
            mix[b * L_ + j] = t0 * t1 * t2;
        }
    }
}

// ---------------- conv1 + 2x2 maxpool + relu --------------------------------------
// N column = pool output; 4 WMMAs per tile (one per 2x2 subpixel) -> pooling is
// in-lane max over accumulators, no shuffles. im2col panels built once per block.
// ypool layout: [(b*8+l)*256 + pooled_pixel][64 ch]  (channel innermost)
__global__ __launch_bounds__(256) void k_conv1_pool(const float* __restrict__ x,
                                                    const _Float16* __restrict__ wf1,
                                                    const float* __restrict__ cb1,
                                                    _Float16* __restrict__ ypool) {
    __shared__ _Float16 sX[C_ * 34 * 34];      // haloed image f16 (~6.9 KB)
    __shared__ _Float16 sB[256 * 32];          // im2col panels [(prow*4+sub)*16+n][32] (16 KB)
    __shared__ _Float16 sW[512 * K1PAD];       // conv1 weights (32 KB)
    __shared__ float    sBias[512];
    int b    = blockIdx.x >> 2;
    int quad = blockIdx.x & 3;                 // pool rows 4*quad .. 4*quad+3
    int tid  = threadIdx.x;

    for (int idx = tid; idx < C_ * 1156; idx += 256) {
        int c = idx / 1156, rem = idx % 1156;
        int yy = rem / 34 - 1, xx = rem % 34 - 1;
        float v = 0.0f;
        if (yy >= 0 && yy < 32 && xx >= 0 && xx < 32)
            v = x[((b * C_ + c) * 32 + yy) * 32 + xx];
        sX[idx] = (_Float16)v;
    }
    __syncthreads();

    // build im2col panels once (gather amortized over all 32 m-tiles)
    for (int idx = tid; idx < 256 * 32; idx += 256) {
        int col = idx >> 5, kk = idx & 31;
        int g = col >> 4, n = col & 15;
        int prow = g >> 2, sub = g & 3;
        int py = 2 * (quad * 4 + prow) + (sub >> 1);
        int px = 2 * n + (sub & 1);
        _Float16 v = (_Float16)0.0f;
        if (kk < 27) {
            int ic = kk / 9, r = kk % 9;
            v = sX[ic * 1156 + (py + r / 3) * 34 + (px + r % 3)];
        }
        sB[idx] = v;
    }
    // stage weights (b128 copies) + bias
    for (int idx = tid; idx < 512 * K1PAD / 8; idx += 256)
        *(v8h*)(sW + idx * 8) = *(const v8h*)(wf1 + idx * 8);
    for (int idx = tid; idx < 512; idx += 256) sBias[idx] = cb1[idx];
    __syncthreads();

    int wave = tid >> 5, lane = tid & 31;
    int m = lane & 15, hs = lane >> 4;
    int n = m;                                 // B: N = lane % 16 = pool column

#pragma unroll 1
    for (int t = wave; t < 128; t += 8) {      // 32 m-tiles x 4 pool rows
        int mtile = t >> 2, prow = t & 3;
        int mo = mtile * 16;

        const _Float16* wrow = sW + (mo + m) * K1PAD + 8 * hs;
        v16h av;
        ((v8h*)&av)[0] = *(const v8h*)wrow;
        ((v8h*)&av)[1] = *(const v8h*)(wrow + 16);

        v8f acc[4];
#pragma unroll
        for (int s = 0; s < 4; ++s) {          // one WMMA per 2x2 subpixel
            const _Float16* sp = sB + ((prow * 4 + s) * 16 + n) * 32 + 8 * hs;
            v16h bv;
            ((v8h*)&bv)[0] = *(const v8h*)sp;
            ((v8h*)&bv)[1] = *(const v8h*)(sp + 16);
            v8f c0 = {};
            acc[s] = __builtin_amdgcn_wmma_f32_16x16x32_f16(false, av, false, bv,
                                                            (short)0, c0, false, false);
        }

        // in-lane 2x2 max + bias + relu; unconditional 16B store per lane
        int pp  = (quad * 4 + prow) * 16 + n;
        int l1  = mo >> 6;
        int ocb = (mo & 63) + 8 * hs;
        v8h pv;
#pragma unroll
        for (int rr = 0; rr < 8; ++rr) {
            float mval = fmaxf(fmaxf(acc[0][rr], acc[1][rr]),
                               fmaxf(acc[2][rr], acc[3][rr]));
            float vv = mval + sBias[mo + rr + 8 * hs];
            pv[rr] = (_Float16)fmaxf(vv, 0.0f);
        }
        *(v8h*)(ypool + ((size_t)(b * L_ + l1) * NPIX2 + pp) * OC_ + ocb) = pv;
    }
}

// ---------------- conv2: 9 shifted GEMMs, K = 64 ic = 2 WMMA chunks -----------------
// LDS: haloed 18x18 tile, channel innermost -> B fragments are contiguous b128 loads.
// Each wave: 1 m-tile x 8 n-tiles (8 accumulators), A fragment amortized over 8 WMMAs.
__global__ __launch_bounds__(256) void k_conv2(const _Float16* __restrict__ ypool,
                                               const _Float16* __restrict__ wf2,
                                               const float* __restrict__ cb2,
                                               _Float16* __restrict__ y2) {
    __shared__ _Float16 sIn[18 * 18 * OC_];            // [pixel_haloed][ic], 41.5 KB
    int b = blockIdx.x >> 3, l = blockIdx.x & 7;
    int tid = threadIdx.x;

    const _Float16* src = ypool + (size_t)(b * L_ + l) * NPIX2 * OC_;
    for (int idx = tid; idx < 324 * 8; idx += 256) {   // 16B chunks
        int pix = idx >> 3, part = idx & 7;
        int yy = pix / 18 - 1, xx = pix % 18 - 1;
        v8h v = {};
        if (yy >= 0 && yy < 16 && xx >= 0 && xx < 16)
            v = *(const v8h*)(src + (yy * 16 + xx) * OC_ + part * 8);
        *(v8h*)(sIn + pix * OC_ + part * 8) = v;
    }
    __syncthreads();

    int wave = tid >> 5, lane = tid & 31;
    int m = lane & 15, hs = lane >> 4;
    int n = m;                                         // B: N = lane % 16
    int mtile = wave & 3;
    int nbase = (wave >> 2) * 8;                       // 8 n-tiles per wave
    int mo = mtile * 16;

    v8f acc[8];
#pragma unroll
    for (int i = 0; i < 8; ++i) acc[i] = (v8f){};

    const _Float16* wbase = wf2 + (size_t)l * 9 * OC_ * OC_ + (mo + m) * OC_;

#pragma unroll 1
    for (int r = 0; r < 9; ++r) {
        int dy = r / 3, dx = r % 3;
        const _Float16* wr = wbase + r * OC_ * OC_;
        if (r < 8) __builtin_prefetch(wbase + (r + 1) * OC_ * OC_, 0, 1);
#pragma unroll
        for (int kc = 0; kc < 2; ++kc) {
            const _Float16* wk = wr + kc * 32;
            v16h av;
            ((v8h*)&av)[0] = *(const v8h*)(wk + 8 * hs);
            ((v8h*)&av)[1] = *(const v8h*)(wk + 16 + 8 * hs);
#pragma unroll
            for (int i = 0; i < 8; ++i) {
                int yy = nbase + i + dy;               // pool row + tap dy, 0..17
                int xx = n + dx;                       // 0..17
                const _Float16* sp = sIn + (yy * 18 + xx) * OC_ + kc * 32 + 8 * hs;
                v16h bv;
                ((v8h*)&bv)[0] = *(const v8h*)sp;
                ((v8h*)&bv)[1] = *(const v8h*)(sp + 16);
                acc[i] = __builtin_amdgcn_wmma_f32_16x16x32_f16(false, av, false, bv,
                                                                (short)0, acc[i],
                                                                false, false);
            }
        }
    }

    // bias + relu + one 16B store per (lane, n-tile); y2 layout [b][l][pix][oc]
#pragma unroll
    for (int i = 0; i < 8; ++i) {
        int ntile = nbase + i;
        v8h ov;
#pragma unroll
        for (int rr = 0; rr < 8; ++rr) {
            int oc = mo + rr + 8 * hs;
            ov[rr] = (_Float16)fmaxf(acc[i][rr] + cb2[l * OC_ + oc], 0.0f);
        }
        int pix = ntile * 16 + n;
        *(v8h*)(y2 + ((size_t)(b * L_ + l) * NPIX2 + pix) * OC_ + mo + 8 * hs) = ov;
    }
}

// ---------------- mixture-weighted leaf sum fused with FC (10 outputs) --------------
// y2 storage order j = pix*64 + oc; reference flatten index i = oc*256 + pix.
__global__ __launch_bounds__(256) void k_final(const _Float16* __restrict__ y2,
                                               const float* __restrict__ mixw,
                                               const float* __restrict__ fcw,
                                               const float* __restrict__ fcb,
                                               float* __restrict__ out) {
    __shared__ float sRed[OUTW * 256];
    int b = blockIdx.x, tid = threadIdx.x;
    float mx[L_];
#pragma unroll
    for (int l = 0; l < L_; ++l) mx[l] = mixw[b * L_ + l];
    float acc[OUTW];
#pragma unroll
    for (int o = 0; o < OUTW; ++o) acc[o] = 0.0f;

    const _Float16* yb = y2 + (size_t)b * L_ * FCIN;
    for (int j = tid; j < FCIN; j += 256) {
        float mixed = 0.0f;
#pragma unroll
        for (int l = 0; l < L_; ++l) mixed += mx[l] * (float)yb[l * FCIN + j];
        int oc = j & 63, pix = j >> 6;
        int i = oc * NPIX2 + pix;                      // reference flatten order
#pragma unroll
        for (int o = 0; o < OUTW; ++o) acc[o] += mixed * fcw[o * FCIN + i];
    }
#pragma unroll
    for (int o = 0; o < OUTW; ++o) sRed[o * 256 + tid] = acc[o];
    __syncthreads();
    if (tid < OUTW) {
        float s = 0.0f;
        for (int t = 0; t < 256; ++t) s += sRed[tid * 256 + t];
        out[b * OUTW + tid] = s + fcb[tid];
    }
}

// ---------------- launch ----------------
extern "C" void kernel_launch(void* const* d_in, const int* in_sizes, int n_in,
                              void* d_out, int out_size, void* d_ws, size_t ws_size,
                              hipStream_t stream) {
    const float* x   = (const float*)d_in[0];
    const float* nw  = (const float*)d_in[1];
    const float* nb  = (const float*)d_in[2];
    const float* w1  = (const float*)d_in[3];
    const float* cb1 = (const float*)d_in[4];
    const float* w2  = (const float*)d_in[5];
    const float* cb2 = (const float*)d_in[6];
    const float* fcw = (const float*)d_in[7];
    const float* fcb = (const float*)d_in[8];
    float* out = (float*)d_out;

    char* ws = (char*)d_ws;
    // workspace layout (bytes):
    //   mix   : 256*8*4        =      8192
    //   wf1   : 512*32*2       =     32768
    //   wf2   : 8*9*64*64*2    =    589824
    //   ypool : 256*8*256*64*2 =  67108864
    //   y2    : 256*8*256*64*2 =  67108864
    float*    mix   = (float*)(ws);
    _Float16* wf1   = (_Float16*)(ws + 8192);
    _Float16* wf2   = (_Float16*)(ws + 8192 + 32768);
    _Float16* ypool = (_Float16*)(ws + 8192 + 32768 + 589824);
    _Float16* y2    = (_Float16*)(ws + 8192 + 32768 + 589824 + 67108864ull);

    k_convert_w1<<<(512 * K1PAD + 255) / 256, 256, 0, stream>>>(w1, wf1);
    k_convert_w2<<<(L_ * 9 * OC_ * OC_ + 255) / 256, 256, 0, stream>>>(w2, wf2);
    k_mixture<<<B_, 256, 0, stream>>>(x, nw, nb, mix);
    k_conv1_pool<<<B_ * 4, 256, 0, stream>>>(x, wf1, cb1, ypool);
    k_conv2<<<B_ * L_, 256, 0, stream>>>(ypool, wf2, cb2, y2);
    k_final<<<B_, 256, 0, stream>>>(y2, mix, fcw, fcb, out);
}
